// MoEFFN_23965917511799
// MI455X (gfx1250) — matrix operand order repaired and checked by reading, not compiled, not run on hardware
//
#include <hip/hip_runtime.h>
#include <hip/hip_bf16.h>

// ---------------- problem constants ----------------
#define N_TOK 8192
#define D_DIM 1024
#define F_DIM 2048
#define E_NUM 8
#define TOPK  2
#define CAPC  2560            // ceil(N*K/E * 1.25)
#define ROWS  32              // token rows per workgroup (2 WMMA row-tiles)
#define FBLK  256             // output-column block (8 waves x 32 cols)
#define KC    32              // WMMA K step (bf16)
#define BPE   (CAPC / ROWS)   // 80 row-blocks per expert

typedef __attribute__((ext_vector_type(16))) __bf16 v16bf;
typedef __attribute__((ext_vector_type(8)))  float  v8f;

union FragU { uint4 q[2]; v16bf v; };

// native bf16 conversion (v_cvt_pk_bf16_f32 on gfx1250, RNE)
__device__ __forceinline__ unsigned short f2bf(float f) {
  __bf16 h = (__bf16)f;
  return __builtin_bit_cast(unsigned short, h);
}
__device__ __forceinline__ unsigned int pk2bf(float a, float b) {
  return (unsigned int)f2bf(a) | ((unsigned int)f2bf(b) << 16);
}

// ---------------- LDS layout (ushort units) ----------------
#define XS_STRIDE (D_DIM + 8)     // 2064B row stride (16B aligned, bank-skewed)
#define ACT_STRIDE (F_DIM + 8)    // 4112B
#define WT_STRIDE 48              // 96B row stride
#define XS_ELEMS  (ROWS * XS_STRIDE)
#define ACT_ELEMS (ROWS * ACT_STRIDE)
#define WT_ELEMS  (FBLK * WT_STRIDE)
#define SMEM_BYTES ((XS_ELEMS + ACT_ELEMS + 4 * WT_ELEMS) * 2 + ROWS * 8)

// ---------------- small kernels ----------------
__global__ void zero_kernel(float* p, size_t n) {
  size_t i  = (size_t)blockIdx.x * blockDim.x + threadIdx.x;
  size_t st = (size_t)gridDim.x * blockDim.x;
  for (; i < n; i += st) p[i] = 0.f;
}

__global__ void init_disp_kernel(int* dtok, float* dw) {
  int i = blockIdx.x * blockDim.x + threadIdx.x;
  if (i < E_NUM * CAPC) { dtok[i] = -1; dw[i] = 0.f; }
}

__global__ void router_kernel(const float* __restrict__ x, const float* __restrict__ Wg,
                              int* __restrict__ top_i, float* __restrict__ top_w,
                              float* __restrict__ counts, float* __restrict__ Psum,
                              float* __restrict__ zsum) {
  int n = blockIdx.x * blockDim.x + threadIdx.x;
  if (n >= N_TOK) return;
  float acc[E_NUM];
#pragma unroll
  for (int e = 0; e < E_NUM; ++e) acc[e] = 0.f;
  const float* xr = x + (size_t)n * D_DIM;
  for (int d = 0; d < D_DIM; ++d) {
    float xv = xr[d];
    const float* wg = Wg + (size_t)d * E_NUM;
#pragma unroll
    for (int e = 0; e < E_NUM; ++e) acc[e] = fmaf(xv, wg[e], acc[e]);
  }
  float m = acc[0];
#pragma unroll
  for (int e = 1; e < E_NUM; ++e) m = fmaxf(m, acc[e]);
  float w[E_NUM]; float s = 0.f;
#pragma unroll
  for (int e = 0; e < E_NUM; ++e) { w[e] = __expf(acc[e] - m); s += w[e]; }
  float inv = 1.f / s;
#pragma unroll
  for (int e = 0; e < E_NUM; ++e) w[e] *= inv;
  float lse = m + __logf(s);
  atomicAdd(zsum, lse * lse);
#pragma unroll
  for (int e = 0; e < E_NUM; ++e) atomicAdd(&Psum[e], w[e]);
  int i1 = 0;
#pragma unroll
  for (int e = 1; e < E_NUM; ++e) if (w[e] > w[i1]) i1 = e;
  int i2 = (i1 == 0) ? 1 : 0;
#pragma unroll
  for (int e = 0; e < E_NUM; ++e) if (e != i1 && w[e] > w[i2]) i2 = e;
  float ts = w[i1] + w[i2];
  top_i[n * 2 + 0] = i1; top_i[n * 2 + 1] = i2;
  top_w[n * 2 + 0] = w[i1] / ts; top_w[n * 2 + 1] = w[i2] / ts;
  atomicAdd(&counts[i1], 1.f); atomicAdd(&counts[i2], 1.f);
}

__global__ void aux_kernel(const float* counts, const float* Psum, const float* zsum,
                           float* out) {
  if (blockIdx.x == 0 && threadIdx.x == 0) {
    float bal = 0.f;
    for (int e = 0; e < E_NUM; ++e)
      bal += (counts[e] / (float)(N_TOK * TOPK)) * (Psum[e] / (float)N_TOK);
    bal *= (float)E_NUM;
    float z = zsum[0] / (float)N_TOK;
    out[(size_t)N_TOK * D_DIM] = 0.01f * bal + 0.001f * z;
  }
}

// Exact top-C-per-expert selection (matches jax top_k ordering: value desc, index asc)
__global__ void dispatch_kernel(const int* __restrict__ top_i, const float* __restrict__ top_w,
                                int* __restrict__ disp_tok, float* __restrict__ disp_w,
                                float* __restrict__ kept_w) {
  int j = blockIdx.x * blockDim.x + threadIdx.x;
  if (j >= N_TOK * TOPK) return;
  int e = top_i[j];
  float wj = top_w[j];
  int rank = 0;
  for (int i = 0; i < N_TOK * TOPK; ++i) {
    if (top_i[i] == e) {
      float wi = top_w[i];
      rank += (wi > wj) || (wi == wj && i < j);
    }
  }
  if (rank < CAPC) {
    disp_tok[e * CAPC + rank] = j >> 1;   // token = j / K
    disp_w[e * CAPC + rank]   = wj;
    kept_w[j] = wj;
  }
}

__global__ void cw_kernel(const float* __restrict__ kept_w, float* __restrict__ cw) {
  int n = blockIdx.x * blockDim.x + threadIdx.x;
  if (n < N_TOK) cw[n] = kept_w[2 * n] + kept_w[2 * n + 1];
}

__global__ void combine_kernel(const float* __restrict__ acc, const float* __restrict__ cw,
                               const float* __restrict__ shared, float* __restrict__ out) {
  size_t i = (size_t)blockIdx.x * blockDim.x + threadIdx.x;
  if (i >= (size_t)N_TOK * D_DIM) return;
  int n = (int)(i / D_DIM);
  float c = cw[n];
  float r = acc[i];
  float v = (c > 0.f) ? r / fmaxf(c, 1e-9f) : r;
  out[i] = v + shared[i];
}

// ---------------- fused SwiGLU FFN (WMMA bf16, f32 accum) ----------------
// Stage a KC x FBLK fp32 weight tile into LDS as bf16 transposed [n][k].
// Each thread owns a 4(k) x 8(n) sub-tile: 8 coalesced float4 loads,
// register transpose, 8 packed b64 LDS stores (compiler merges to 2addr_b64).
__device__ __forceinline__ void stage_wtile(const float* __restrict__ src, size_t ld,
                                            unsigned short* __restrict__ dst, int tid,
                                            bool prefetch_next, size_t next_off) {
  int k0 = (tid >> 5) * 4;
  int n0 = (tid & 31) * 8;
  const float* s = src + (size_t)k0 * ld + n0;
  float4 ra[4], rb[4];
#pragma unroll
  for (int i = 0; i < 4; ++i) {
    ra[i] = *(const float4*)(s + (size_t)i * ld);
    rb[i] = *(const float4*)(s + (size_t)i * ld + 4);
  }
  if (prefetch_next) __builtin_prefetch(s + next_off, 0, 1);
#pragma unroll
  for (int j = 0; j < 4; ++j) {
    uint2 p;
    p.x = pk2bf((&ra[0].x)[j], (&ra[1].x)[j]);
    p.y = pk2bf((&ra[2].x)[j], (&ra[3].x)[j]);
    *(uint2*)(dst + (n0 + j) * WT_STRIDE + k0) = p;
    uint2 q;
    q.x = pk2bf((&rb[0].x)[j], (&rb[1].x)[j]);
    q.y = pk2bf((&rb[2].x)[j], (&rb[3].x)[j]);
    *(uint2*)(dst + (n0 + 4 + j) * WT_STRIDE + k0) = q;
  }
}

__device__ __forceinline__ void load_bfrag(FragU& f, const unsigned short* base,
                                           int row, int lh) {
  const unsigned short* bp = base + row * WT_STRIDE + lh * 8;
  f.q[0] = *(const uint4*)(bp);
  f.q[1] = *(const uint4*)(bp + 16);
}

// One K-step of GEMM1 (two B matrices share the A fragments).
__device__ __forceinline__ void gemm1_step(const unsigned short* __restrict__ Xs,
                                           const unsigned short* __restrict__ Wac,
                                           const unsigned short* __restrict__ Wbc,
                                           int kb, int cbase, int lm, int lh,
                                           v8f (&h1)[2][2], v8f (&h3)[2][2]) {
  FragU afr[2];
#pragma unroll
  for (int rt = 0; rt < 2; ++rt) {
    const unsigned short* ap = Xs + (rt * 16 + lm) * XS_STRIDE + kb + lh * 8;
    afr[rt].q[0] = *(const uint4*)(ap);
    afr[rt].q[1] = *(const uint4*)(ap + 16);
  }
#pragma unroll
  for (int ct = 0; ct < 2; ++ct) {
    FragU b1, b3;
    load_bfrag(b1, Wac, cbase + ct * 16 + lm, lh);
    load_bfrag(b3, Wbc, cbase + ct * 16 + lm, lh);
#pragma unroll
    for (int rt = 0; rt < 2; ++rt) {
      h1[rt][ct] = __builtin_amdgcn_wmma_f32_16x16x32_bf16(
          false, afr[rt].v, false, b1.v, (short)0, h1[rt][ct], false, false);
      h3[rt][ct] = __builtin_amdgcn_wmma_f32_16x16x32_bf16(
          false, afr[rt].v, false, b3.v, (short)0, h3[rt][ct], false, false);
    }
  }
}

// One K-step of GEMM2.
__device__ __forceinline__ void gemm2_step(const unsigned short* __restrict__ Act,
                                           const unsigned short* __restrict__ Wc,
                                           int kb, int cbase, int lm, int lh,
                                           v8f (&y)[2][2]) {
  FragU afr[2];
#pragma unroll
  for (int rt = 0; rt < 2; ++rt) {
    const unsigned short* ap = Act + (rt * 16 + lm) * ACT_STRIDE + kb + lh * 8;
    afr[rt].q[0] = *(const uint4*)(ap);
    afr[rt].q[1] = *(const uint4*)(ap + 16);
  }
#pragma unroll
  for (int ct = 0; ct < 2; ++ct) {
    FragU b;
    load_bfrag(b, Wc, cbase + ct * 16 + lm, lh);
#pragma unroll
    for (int rt = 0; rt < 2; ++rt) {
      y[rt][ct] = __builtin_amdgcn_wmma_f32_16x16x32_bf16(
          false, afr[rt].v, false, b.v, (short)0, y[rt][ct], false, false);
    }
  }
}

template <bool ROUTED>
__global__ __launch_bounds__(256)
void ffn_kernel(const float* __restrict__ X,
                const float* __restrict__ W1, const float* __restrict__ W3,
                const float* __restrict__ W2,
                const int* __restrict__ disp_tok, const float* __restrict__ disp_w,
                float* __restrict__ out) {
  extern __shared__ char smem[];
  unsigned short* Xs  = (unsigned short*)smem;
  unsigned short* Act = Xs + XS_ELEMS;
  unsigned short* Wt  = Act + ACT_ELEMS;   // 4 x WT_ELEMS; offsets are literals below
  int*   tokS = (int*)(Wt + 4 * WT_ELEMS);
  float* wS   = (float*)(tokS + ROWS);

  const int tid = threadIdx.x;
  const float *W1e, *W3e, *W2e;
  int slot0 = 0, tokbase = 0;
  if (ROUTED) {
    int e  = blockIdx.x / BPE;
    int rb = blockIdx.x % BPE;
    slot0 = e * CAPC + rb * ROWS;
    W1e = W1 + (size_t)e * D_DIM * F_DIM;
    W3e = W3 + (size_t)e * D_DIM * F_DIM;
    W2e = W2 + (size_t)e * F_DIM * D_DIM;
  } else {
    tokbase = blockIdx.x * ROWS;
    W1e = W1; W3e = W3; W2e = W2;
  }
  if (tid < ROWS) {
    if (ROUTED) { tokS[tid] = disp_tok[slot0 + tid]; wS[tid] = disp_w[slot0 + tid]; }
    else        { tokS[tid] = tokbase + tid;          wS[tid] = 1.0f; }
  }
  __syncthreads();

  // gather 32 token rows -> LDS bf16 (packed b64 stores)
  {
    int r  = tid >> 3;
    int c0 = (tid & 7) * 128;
    int tok = tokS[r];
    unsigned short* dst = Xs + r * XS_STRIDE + c0;
    if (tok >= 0) {
      const float4* src = (const float4*)(X + (size_t)tok * D_DIM + c0);
#pragma unroll 4
      for (int i = 0; i < 32; ++i) {
        float4 v = src[i];
        uint2 p; p.x = pk2bf(v.x, v.y); p.y = pk2bf(v.z, v.w);
        *(uint2*)(dst + 4 * i) = p;
      }
    } else {
      uint2 z; z.x = 0; z.y = 0;
      for (int i = 0; i < 32; ++i) *(uint2*)(dst + 4 * i) = z;
    }
  }
  __syncthreads();

  const int wv    = tid >> 5;
  const int lane  = tid & 31;
  const int cbase = wv * 32;                // 32 output cols per wave within FBLK
  const int lm    = lane & 15;
  const int lh    = lane >> 4;
  const v8f vzero = {0.f, 0.f, 0.f, 0.f, 0.f, 0.f, 0.f, 0.f};

  // -------- GEMM1: H1 = X@W1, H3 = X@W3, act = silu(H1)*H3 --------
  for (int fb = 0; fb < F_DIM; fb += FBLK) {
    v8f h1[2][2], h3[2][2];                 // [row-tile][col-tile]
#pragma unroll
    for (int rt = 0; rt < 2; ++rt)
#pragma unroll
      for (int ct = 0; ct < 2; ++ct) { h1[rt][ct] = vzero; h3[rt][ct] = vzero; }

    // prologue: stage first K-tile into buffer 0 (literal LDS offsets)
    stage_wtile(W1e + fb, F_DIM, Wt + 0 * WT_ELEMS, tid, true, (size_t)KC * F_DIM);
    stage_wtile(W3e + fb, F_DIM, Wt + 1 * WT_ELEMS, tid, true, (size_t)KC * F_DIM);
    __syncthreads();

    // K loop unrolled x2 so buffer selection is compile-time constant
    for (int kb = 0; kb < D_DIM; kb += 2 * KC) {
      {  // phase 0: read buffer 0 (tile kb), stage buffer 1 (tile kb+KC)
        bool pf = (kb + 2 * KC) < D_DIM;
        stage_wtile(W1e + (size_t)(kb + KC) * F_DIM + fb, F_DIM,
                    Wt + 2 * WT_ELEMS, tid, pf, (size_t)KC * F_DIM);
        stage_wtile(W3e + (size_t)(kb + KC) * F_DIM + fb, F_DIM,
                    Wt + 3 * WT_ELEMS, tid, pf, (size_t)KC * F_DIM);
        gemm1_step(Xs, Wt + 0 * WT_ELEMS, Wt + 1 * WT_ELEMS, kb, cbase, lm, lh, h1, h3);
        __syncthreads();
      }
      {  // phase 1: read buffer 1 (tile kb+KC), stage buffer 0 (tile kb+2KC)
        if (kb + 2 * KC < D_DIM) {
          bool pf = (kb + 3 * KC) < D_DIM;
          stage_wtile(W1e + (size_t)(kb + 2 * KC) * F_DIM + fb, F_DIM,
                      Wt + 0 * WT_ELEMS, tid, pf, (size_t)KC * F_DIM);
          stage_wtile(W3e + (size_t)(kb + 2 * KC) * F_DIM + fb, F_DIM,
                      Wt + 1 * WT_ELEMS, tid, pf, (size_t)KC * F_DIM);
        }
        gemm1_step(Xs, Wt + 2 * WT_ELEMS, Wt + 3 * WT_ELEMS, kb + KC, cbase, lm, lh, h1, h3);
        __syncthreads();
      }
    }
    // activation: silu(h1) * h3 -> Act (bf16); fast rcp instead of IEEE divide
#pragma unroll
    for (int rt = 0; rt < 2; ++rt) {
#pragma unroll
      for (int ct = 0; ct < 2; ++ct) {
#pragma unroll
        for (int r = 0; r < 8; ++r) {
          float a = h1[rt][ct][r];
          float sig = __builtin_amdgcn_rcpf(1.f + __expf(-a));
          float v = a * sig * h3[rt][ct][r];
          int row = rt * 16 + r + 8 * lh;
          int col = fb + cbase + ct * 16 + lm;
          Act[row * ACT_STRIDE + col] = f2bf(v);
        }
      }
    }
  }
  __syncthreads();

  // -------- GEMM2: Y = Act @ W2, epilogue scatter --------
  for (int cb = 0; cb < D_DIM; cb += FBLK) {
    v8f y[2][2];
#pragma unroll
    for (int rt = 0; rt < 2; ++rt)
#pragma unroll
      for (int ct = 0; ct < 2; ++ct) y[rt][ct] = vzero;

    stage_wtile(W2e + cb, D_DIM, Wt + 0 * WT_ELEMS, tid, true, (size_t)KC * D_DIM);
    __syncthreads();

    for (int kb = 0; kb < F_DIM; kb += 2 * KC) {
      {  // phase 0: read buffer 0, stage buffer 1
        bool pf = (kb + 2 * KC) < F_DIM;
        stage_wtile(W2e + (size_t)(kb + KC) * D_DIM + cb, D_DIM,
                    Wt + 1 * WT_ELEMS, tid, pf, (size_t)KC * D_DIM);
        gemm2_step(Act, Wt + 0 * WT_ELEMS, kb, cbase, lm, lh, y);
        __syncthreads();
      }
      {  // phase 1: read buffer 1, stage buffer 0
        if (kb + 2 * KC < F_DIM) {
          bool pf = (kb + 3 * KC) < F_DIM;
          stage_wtile(W2e + (size_t)(kb + 2 * KC) * D_DIM + cb, D_DIM,
                      Wt + 0 * WT_ELEMS, tid, pf, (size_t)KC * D_DIM);
        }
        gemm2_step(Act, Wt + 1 * WT_ELEMS, kb + KC, cbase, lm, lh, y);
        __syncthreads();
      }
    }
#pragma unroll
    for (int rt = 0; rt < 2; ++rt) {
#pragma unroll
      for (int ct = 0; ct < 2; ++ct) {
#pragma unroll
        for (int r = 0; r < 8; ++r) {
          int row = rt * 16 + r + 8 * lh;
          int tok = tokS[row];
          if (tok >= 0) {
            int col = cb + cbase + ct * 16 + lm;
            float v = y[rt][ct][r];
            if (ROUTED) atomicAdd(out + (size_t)tok * D_DIM + col, v * wS[row]);
            else        out[(size_t)tok * D_DIM + col] = v;
          }
        }
      }
    }
  }
}

// ---------------- host launcher ----------------
extern "C" void kernel_launch(void* const* d_in, const int* in_sizes, int n_in,
                              void* d_out, int out_size, void* d_ws, size_t ws_size,
                              hipStream_t stream) {
  (void)in_sizes; (void)n_in; (void)out_size; (void)ws_size;
  const float* x   = (const float*)d_in[0];
  const float* Wg  = (const float*)d_in[1];
  const float* W1  = (const float*)d_in[2];
  const float* W3  = (const float*)d_in[3];
  const float* W2  = (const float*)d_in[4];
  const float* Ws1 = (const float*)d_in[5];
  const float* Ws3 = (const float*)d_in[6];
  const float* Ws2 = (const float*)d_in[7];
  float* out = (float*)d_out;

  float* ws = (float*)d_ws;
  size_t o_acc = 0;                                   // N*D   routed accumulator
  size_t o_cw  = o_acc + (size_t)N_TOK * D_DIM;       // N     combine weights
  size_t o_cnt = o_cw + N_TOK;                        // E     counts
  size_t o_ps  = o_cnt + E_NUM;                       // E     softmax-prob sums
  size_t o_zs  = o_ps + E_NUM;                        // 1(+7) z-loss sum
  size_t o_kw  = o_zs + 8;                            // N*K   kept weights
  size_t o_tw  = o_kw + (size_t)N_TOK * TOPK;         // N*K   top_w
  size_t o_dw  = o_tw + (size_t)N_TOK * TOPK;         // E*C   dispatch weights
  size_t o_ti  = o_dw + (size_t)E_NUM * CAPC;         // N*K   top_i (int)
  size_t o_dt  = o_ti + (size_t)N_TOK * TOPK;         // E*C   dispatch tokens (int)
  size_t o_sh  = o_dt + (size_t)E_NUM * CAPC;         // N*D   shared-expert output

  float* acc = ws + o_acc; float* cw = ws + o_cw; float* cnt = ws + o_cnt;
  float* ps  = ws + o_ps;  float* zs = ws + o_zs; float* kw  = ws + o_kw;
  float* tw  = ws + o_tw;  float* dw = ws + o_dw;
  int*   ti  = (int*)(ws + o_ti);
  int*   dt  = (int*)(ws + o_dt);
  float* sh  = ws + o_sh;

  (void)hipFuncSetAttribute(reinterpret_cast<const void*>(ffn_kernel<true>),
                            hipFuncAttributeMaxDynamicSharedMemorySize, SMEM_BYTES);
  (void)hipFuncSetAttribute(reinterpret_cast<const void*>(ffn_kernel<false>),
                            hipFuncAttributeMaxDynamicSharedMemorySize, SMEM_BYTES);

  size_t zn = o_kw + (size_t)N_TOK * TOPK;  // zero [acc .. kept_w] contiguous region
  zero_kernel<<<4096, 256, 0, stream>>>(ws, zn);
  init_disp_kernel<<<(E_NUM * CAPC + 255) / 256, 256, 0, stream>>>(dt, dw);
  router_kernel<<<N_TOK / 256, 256, 0, stream>>>(x, Wg, ti, tw, cnt, ps, zs);
  aux_kernel<<<1, 1, 0, stream>>>(cnt, ps, zs, out);
  dispatch_kernel<<<(N_TOK * TOPK) / 256, 256, 0, stream>>>(ti, tw, dt, dw, kw);
  cw_kernel<<<N_TOK / 256, 256, 0, stream>>>(kw, cw);
  ffn_kernel<true><<<E_NUM * BPE, 256, SMEM_BYTES, stream>>>(x, W1, W3, W2, dt, dw, acc);
  ffn_kernel<false><<<N_TOK / ROWS, 256, SMEM_BYTES, stream>>>(x, Ws1, Ws3, Ws2,
                                                               nullptr, nullptr, sh);
  combine_kernel<<<(N_TOK * D_DIM) / 256, 256, 0, stream>>>(acc, cw, sh, out);
}
